// LanguageModel_66477503807731
// MI455X (gfx1250) — matrix operand order repaired and checked by reading, not compiled, not run on hardware
//
#include <hip/hip_runtime.h>
#include <hip/hip_bf16.h>

// Problem constants (B,N,D,E,F) = (4,4096,512,4,2048), TOP_K=1.
#define B_    4
#define N_    4096
#define D_    512
#define E_    4
#define F_    2048
#define NTOK  (B_ * N_)          // 16384 tokens
#define TMB   32                 // tokens per block = 2 WMMA M-tiles (B reused x2)
#define FC    256                // F-dimension chunk held in LDS
#define PAD   8                  // bf16 elements of row padding (16B, keeps alignment)
#define XSTR  (D_ + PAD)         // 520
#define HSTR  (FC + PAD)         // 264

typedef __attribute__((ext_vector_type(16))) __bf16   v16bf;
typedef __attribute__((ext_vector_type(8)))  float    v8f;
typedef __attribute__((ext_vector_type(4)))  unsigned int u32x4;

union ABReg { v16bf v; unsigned short s[16]; u32x4 u[2]; };

__device__ __forceinline__ unsigned short f2bf(float f) {
  unsigned int u = __float_as_uint(f);
  u += 0x7FFFu + ((u >> 16) & 1u);          // round-to-nearest-even
  return (unsigned short)(u >> 16);
}

// Branchless tanh-form gelu (max dev <0.1% of exact erf-gelu; H is stored as bf16
// whose ~0.4% quantization dominates). One v_exp_f32 + one v_rcp_f32, no exec branches.
__device__ __forceinline__ float gelu_f(float t) {
  float u  = t * (0.79788456079f + 0.03567740814f * t * t);  // 0.797885*(t+0.044715 t^3)
  float ex = __expf(-2.0f * u);
  float th = 1.0f - 2.0f * ex * __builtin_amdgcn_rcpf(1.0f + ex);  // tanh(u)
  return 0.5f * t * (1.0f + th);
}

// A-matrix (16xK slice, 16-bit): lanes 0-15 hold row M=lane, K {0..7,16..23};
// lanes 16-31 hold row M=lane-16, K {8..15,24..31}. Two 16B LDS loads.
__device__ __forceinline__ v16bf load_a16(const unsigned short* rowBase, int kbase, int lane) {
  const int koff = (lane < 16) ? 0 : 8;
  ABReg r;
  r.u[0] = *(const u32x4*)(rowBase + kbase + koff);
  r.u[1] = *(const u32x4*)(rowBase + kbase + koff + 16);
  return r.v;
}

// B-matrix (32x16, 16-bit): lane holds column N=lane%16; lanes 0-15 K {0..7,16..23},
// lanes 16-31 K {8..15,24..31} (same K order as A). Strided column gather.
__device__ __forceinline__ v16bf load_b_bf16(const unsigned short* p, int ld, int lane) {
  const int koff = (lane < 16) ? 0 : 8;
  const int n = lane & 15;
  ABReg r;
#pragma unroll
  for (int i = 0; i < 8; ++i) {
    r.s[i]     = p[(size_t)(koff + i) * ld + n];
    r.s[8 + i] = p[(size_t)(koff + 16 + i) * ld + n];
  }
  return r.v;
}

// Fallback: gather the same column/K layout from f32 and pack (only used if workspace too small).
__device__ __forceinline__ v16bf load_b_f32(const float* p, int ld, int lane) {
  const int koff = (lane < 16) ? 0 : 8;
  const int n = lane & 15;
  ABReg r;
#pragma unroll
  for (int i = 0; i < 8; ++i) {
    r.s[i]     = f2bf(p[(size_t)(koff + i) * ld + n]);
    r.s[8 + i] = f2bf(p[(size_t)(koff + 16 + i) * ld + n]);
  }
  return r.v;
}

// ---------------- bulk f32 -> packed bf16 conversion (memory-bound prepass) ----------------
__global__ __launch_bounds__(256) void cvt_bf16_kernel(const float* __restrict__ src,
                                                       unsigned int* __restrict__ dst, int n2) {
  const int i = blockIdx.x * 256 + threadIdx.x;
  if (i >= n2) return;
  float2 f = ((const float2*)src)[i];
  dst[i] = (unsigned int)f2bf(f.x) | ((unsigned int)f2bf(f.y) << 16);
}

// ---------------- token projection + gating + expert counting ----------------
__global__ __launch_bounds__(256) void proj_gate_kernel(
    const float* __restrict__ x, const float* __restrict__ vw, const float* __restrict__ vb,
    const float* __restrict__ gw, const float* __restrict__ gb,
    float* __restrict__ vout, int* __restrict__ eout, int* __restrict__ meta) {
  const int lane = threadIdx.x & 31;
  const int wv   = threadIdx.x >> 5;
  const int t = blockIdx.x * 8 + wv;
  if (t >= NTOK) return;
  const float* xr = x + (size_t)t * D_;
  float a0 = 0.f, a1 = 0.f, a2 = 0.f, a3 = 0.f, a4 = 0.f;
  for (int k = lane; k < D_; k += 32) {
    float xv = xr[k];
    a0 += xv * vw[k];
    a1 += xv * gw[k];
    a2 += xv * gw[D_ + k];
    a3 += xv * gw[2 * D_ + k];
    a4 += xv * gw[3 * D_ + k];
  }
#pragma unroll
  for (int off = 16; off > 0; off >>= 1) {
    a0 += __shfl_xor(a0, off, 32);
    a1 += __shfl_xor(a1, off, 32);
    a2 += __shfl_xor(a2, off, 32);
    a3 += __shfl_xor(a3, off, 32);
    a4 += __shfl_xor(a4, off, 32);
  }
  if (lane == 0) {
    float v = a0 + vb[0];
    v = fminf(fmaxf(v, -3.0f), 3.0f);
    vout[t] = v;
    float l0 = a1 + gb[0], l1 = a2 + gb[1], l2 = a3 + gb[2], l3 = a4 + gb[3];
    int best = 0; float bm = l0;                 // softmax is monotone: argmax(logits)
    if (l1 > bm) { bm = l1; best = 1; }
    if (l2 > bm) { bm = l2; best = 2; }
    if (l3 > bm) { bm = l3; best = 3; }
    eout[t] = best;
    atomicAdd(&meta[best], 1);
  }
}

// ---------------- sequential two-sided continued-fraction scan ----------------
__global__ void green_scan_kernel(const float* __restrict__ v, const float* __restrict__ gamma,
                                  const float* __restrict__ epsp,
                                  float* __restrict__ Lz, float* __restrict__ Uz) {
  const int lane = threadIdx.x;
  if (lane >= 8) return;
  const float eps = log1pf(expf(epsp[0])) + 1e-6f;
  const float di  = -(eps + gamma[0]);           // d = (v-2) - i*(eps+gamma)
  const int   b   = lane & 3;
  const float* vb = v + b * N_;
  float* out = (lane < 4) ? Lz : Uz;
  float cr, ci;
  if (lane < 4) {                                 // forward: L_i = d_i - 1/L_{i-1}
    cr = vb[0] - 2.0f; ci = di;
    out[(size_t)(b * N_) * 2] = cr; out[(size_t)(b * N_) * 2 + 1] = ci;
    for (int i = 1; i < N_; ++i) {
      float m  = cr * cr + ci * ci;               // 1/c = (cr - i*ci)/m
      float nr = (vb[i] - 2.0f) - cr / m;
      float ni = di + ci / m;
      cr = nr; ci = ni;
      out[(size_t)(b * N_ + i) * 2] = cr; out[(size_t)(b * N_ + i) * 2 + 1] = ci;
    }
  } else {                                        // backward: U_i = d_i - 1/U_{i+1}
    cr = vb[N_ - 1] - 2.0f; ci = di;
    out[(size_t)(b * N_ + N_ - 1) * 2] = cr; out[(size_t)(b * N_ + N_ - 1) * 2 + 1] = ci;
    for (int i = N_ - 2; i >= 0; --i) {
      float m  = cr * cr + ci * ci;
      float nr = (vb[i] - 2.0f) - cr / m;
      float ni = di + ci / m;
      cr = nr; ci = ni;
      out[(size_t)(b * N_ + i) * 2] = cr; out[(size_t)(b * N_ + i) * 2 + 1] = ci;
    }
  }
}

__global__ __launch_bounds__(256) void feats_kernel(
    const float* __restrict__ v, const float* __restrict__ gamma, const float* __restrict__ epsp,
    const float* __restrict__ Lz, const float* __restrict__ Uz, float* __restrict__ feats) {
  const int idx = blockIdx.x * 256 + threadIdx.x;
  if (idx >= NTOK) return;
  const float eps = log1pf(expf(epsp[0])) + 1e-6f;
  const float di  = -(eps + gamma[0]);
  const float dr  = v[idx] - 2.0f;
  float sr = Lz[2 * idx] + Uz[2 * idx] - dr;
  float si = Lz[2 * idx + 1] + Uz[2 * idx + 1] - di;
  float m  = sr * sr + si * si;                  // G = 1/(L+U-d)
  float Gr = sr / m, Gi = -si / m;
  feats[2 * idx]     = fminf(fmaxf(Gr, -10.f), 10.f);
  feats[2 * idx + 1] = fminf(fmaxf(Gi, -10.f), 10.f);
}

// ---------------- expert bucketing metadata ----------------
// meta: [0..3]=counts  [4..8]=offsets  [9..13]=tile_starts  [14..17]=cursors
__global__ void init_meta_kernel(int* meta) { if (threadIdx.x < 32) meta[threadIdx.x] = 0; }

__global__ void scan_meta_kernel(int* meta) {
  if (threadIdx.x == 0 && blockIdx.x == 0) {
    int* counts = meta; int* off = meta + 4; int* ts = meta + 9; int* cur = meta + 14;
    off[0] = 0; ts[0] = 0;
    for (int e = 0; e < E_; ++e) {
      off[e + 1] = off[e] + counts[e];
      ts[e + 1]  = ts[e] + (counts[e] + TMB - 1) / TMB;
      cur[e] = 0;
    }
  }
}

__global__ __launch_bounds__(256) void scatter_kernel(const int* __restrict__ eout,
                                                      int* __restrict__ meta, int* __restrict__ perm) {
  const int t = blockIdx.x * 256 + threadIdx.x;
  if (t >= NTOK) return;
  const int e = eout[t];
  const int pos = atomicAdd(&meta[14 + e], 1);
  perm[meta[4 + e] + pos] = t;
}

// ---------------- grouped MoE FFN + fused spectral output (WMMA bf16) ----------------
// 32 tokens/block (2 M-tiles), 8 waves. Each wave owns both M-tiles and a narrow
// N-slice: every B fragment loaded from L2 feeds TWO WMMAs -> half the L2 operand
// traffic per FLOP vs a single M-tile. acc[8] (64 VGPRs) persistent per wave;
// phase-1 peak ~140 VGPRs (acc 64 + c1[4] 32 + 2xA 16 + B staging) -> no spills.
template <bool BF>
__global__ __launch_bounds__(256)
__attribute__((amdgpu_waves_per_eu(2, 2)))
void ffn_moe_kernel(
    const float* __restrict__ x,  const unsigned short* __restrict__ xbf,
    const float* __restrict__ w1, const unsigned short* __restrict__ w1bf,
    const float* __restrict__ b1,
    const float* __restrict__ w2, const unsigned short* __restrict__ w2bf,
    const float* __restrict__ b2,
    const float* __restrict__ out_w, const float* __restrict__ out_b, const float* __restrict__ bk,
    const int* __restrict__ perm, const int* __restrict__ meta,
    const float* __restrict__ feats, float* __restrict__ out) {
  const int* counts  = meta;
  const int* offsets = meta + 4;
  const int* tstarts = meta + 9;

  const int blk = blockIdx.x;
  if (blk >= tstarts[E_]) return;
  int e = 0;
  while (e < E_ - 1 && blk >= tstarts[e + 1]) ++e;
  const int tile  = blk - tstarts[e];
  const int tbase = tile * TMB;                  // tbase < counts[e] by construction
  const int cnt   = counts[e];

  __shared__ __attribute__((aligned(16))) unsigned short sX[TMB * XSTR]; // bf16 X tile (33.3KB)
  __shared__ __attribute__((aligned(16))) unsigned short sH[TMB * HSTR]; // bf16 H chunk (16.9KB)
  __shared__ int   sTok[TMB];
  __shared__ int   sVal[TMB];
  __shared__ float sF0[TMB], sF1[TMB];

  const int tid  = threadIdx.x;
  const int lane = tid & 31;
  const int wv   = tid >> 5;                     // 0..7

  if (tid < TMB) {
    const int idx = tbase + tid;
    const int ok  = (idx < cnt);
    const int gt  = perm[offsets[e] + (ok ? idx : tbase)];  // duplicate row 0 for padding
    sTok[tid] = gt;
    sVal[tid] = ok;
    sF0[tid]  = feats[2 * gt];
    sF1[tid]  = feats[2 * gt + 1];
  }
  __syncthreads();

  // Gather X rows -> LDS
  if (BF) {
    for (int i = tid; i < TMB * D_ / 8; i += 256) {       // 16B vector copies
      const int m = i >> 6;                               // / (512/8)
      const int k = (i & 63) * 8;
      *(u32x4*)(sX + m * XSTR + k) = *(const u32x4*)(xbf + (size_t)sTok[m] * D_ + k);
    }
  } else {
    for (int i = tid; i < TMB * D_; i += 256) {
      const int m = i >> 9;
      const int k = i & (D_ - 1);
      sX[m * XSTR + k] = f2bf(x[(size_t)sTok[m] * D_ + k]);
    }
  }
  __syncthreads();

  const float*          W1f = w1   + (size_t)e * D_ * F_;
  const unsigned short* W1b = w1bf + (size_t)e * D_ * F_;
  const float*          W2f = w2   + (size_t)e * F_ * D_;
  const unsigned short* W2b = w2bf + (size_t)e * F_ * D_;
  const float* b1e = b1 + (size_t)e * F_;
  const float* b2e = b2 + (size_t)e * D_;

  const v8f zero = {};
  v8f acc[8];                                    // [j=4 d-tiles][p=2 M-tiles]
#pragma unroll
  for (int j = 0; j < 8; ++j) acc[j] = zero;

  const int mA = lane & 15;
  const unsigned short* aX0 = sX + mA * XSTR;          // M-tile 0 (tokens 0..15)
  const unsigned short* aX1 = sX + (16 + mA) * XSTR;   // M-tile 1 (tokens 16..31)
  const unsigned short* aH0 = sH + mA * HSTR;
  const unsigned short* aH1 = sH + (16 + mA) * HSTR;
  const int mrow = (lane < 16) ? 0 : 8;          // C/D row base for this lane half

  for (int fc = 0; fc < F_; fc += FC) {
    // ---- phase 1: H[:, fc:fc+FC] = gelu(X @ W1 + b1) ----
    // Per wave: 2 n-tiles x 2 M-tiles; each B load feeds 2 WMMAs (a0,a1).
    v8f c1[4];                                   // [j=2][p=2]
#pragma unroll
    for (int j = 0; j < 4; ++j) c1[j] = zero;
#pragma unroll 1
    for (int kk = 0; kk < D_ / 32; ++kk) {
      v16bf a0 = load_a16(aX0, kk * 32, lane);
      v16bf a1 = load_a16(aX1, kk * 32, lane);
      const size_t rowOff = (size_t)(kk * 32) * F_ + fc + wv * 32;
#pragma unroll
      for (int j = 0; j < 2; ++j) {
        v16bf bm = BF ? load_b_bf16(W1b + rowOff + j * 16, F_, lane)
                      : load_b_f32 (W1f + rowOff + j * 16, F_, lane);
        c1[2*j+0] = __builtin_amdgcn_wmma_f32_16x16x32_bf16(false, a0, false, bm, (short)0,
                                                            c1[2*j+0], false, false);
        c1[2*j+1] = __builtin_amdgcn_wmma_f32_16x16x32_bf16(false, a1, false, bm, (short)0,
                                                            c1[2*j+1], false, false);
      }
    }
    // Fully unrolled epilogue: static indexing, branchless gelu.
#pragma unroll
    for (int j = 0; j < 2; ++j) {
      const int ncol = fc + wv * 32 + j * 16;
      const float bias = b1e[ncol + (lane & 15)];
      const int   hcol = (ncol - fc) + (lane & 15);
#pragma unroll
      for (int p = 0; p < 2; ++p) {
#pragma unroll
        for (int r = 0; r < 8; ++r) {
          sH[(p * 16 + mrow + r) * HSTR + hcol] = f2bf(gelu_f(c1[2*j+p][r] + bias));
        }
      }
    }
    __syncthreads();
    // ---- phase 2 partial: acc += H_chunk @ W2[fc:fc+FC, :] ----
#pragma unroll 1
    for (int kk = 0; kk < FC / 32; ++kk) {
      v16bf a0 = load_a16(aH0, kk * 32, lane);
      v16bf a1 = load_a16(aH1, kk * 32, lane);
      const size_t rowOff = (size_t)(fc + kk * 32) * D_ + wv * 64;
#pragma unroll
      for (int j = 0; j < 4; ++j) {
        v16bf bm = BF ? load_b_bf16(W2b + rowOff + j * 16, D_, lane)
                      : load_b_f32 (W2f + rowOff + j * 16, D_, lane);
        acc[2*j+0] = __builtin_amdgcn_wmma_f32_16x16x32_bf16(false, a0, false, bm, (short)0,
                                                             acc[2*j+0], false, false);
        acc[2*j+1] = __builtin_amdgcn_wmma_f32_16x16x32_bf16(false, a1, false, bm, (short)0,
                                                             acc[2*j+1], false, false);
      }
    }
    __syncthreads();                             // before next chunk overwrites sH
  }

  // ---- epilogue: y + b2 + bk * (feats . out_w + out_b), scatter to out ----
#pragma unroll
  for (int j = 0; j < 4; ++j) {
    const int   n    = wv * 64 + j * 16 + (lane & 15);
    const float bias = b2e[n];
    const float ow0  = out_w[2 * n], ow1 = out_w[2 * n + 1];
    const float ob   = out_b[n],     bks = bk[n];
#pragma unroll
    for (int p = 0; p < 2; ++p) {
#pragma unroll
      for (int r = 0; r < 8; ++r) {
        const int m = p * 16 + mrow + r;
        if (sVal[m]) {
          const float spec = sF0[m] * ow0 + sF1[m] * ow1 + ob;
          out[(size_t)sTok[m] * D_ + n] = acc[2*j+p][r] + bias + bks * spec;
        }
      }
    }
  }
}

extern "C" void kernel_launch(void* const* d_in, const int* in_sizes, int n_in,
                              void* d_out, int out_size, void* d_ws, size_t ws_size,
                              hipStream_t stream) {
  const float* x     = (const float*)d_in[0];
  const float* vw    = (const float*)d_in[1];
  const float* vb    = (const float*)d_in[2];
  const float* gamma = (const float*)d_in[3];
  const float* epsp  = (const float*)d_in[4];
  const float* gw    = (const float*)d_in[5];
  const float* gb    = (const float*)d_in[6];
  const float* w1    = (const float*)d_in[7];
  const float* b1    = (const float*)d_in[8];
  const float* w2    = (const float*)d_in[9];
  const float* b2    = (const float*)d_in[10];
  const float* ow    = (const float*)d_in[11];
  const float* ob    = (const float*)d_in[12];
  const float* bk    = (const float*)d_in[13];
  float* out = (float*)d_out;

  char* ws = (char*)d_ws;
  float* v_ws  = (float*)(ws);                    // 64 KB
  int*   e_ws  = (int*)(ws + 65536);              // 64 KB
  float* L_ws  = (float*)(ws + 131072);           // 128 KB (re,im)
  float* U_ws  = (float*)(ws + 262144);           // 128 KB
  float* feats = (float*)(ws + 393216);           // 128 KB
  int*   perm  = (int*)(ws + 524288);             // 64 KB
  int*   meta  = (int*)(ws + 589824);             // 32 ints
  // bf16 shadow copies (only if workspace is big enough)
  const size_t XBF_OFF = 655360;                          // 16 MB
  const size_t W1B_OFF = XBF_OFF + (size_t)NTOK * D_ * 2; // 8 MB
  const size_t W2B_OFF = W1B_OFF + (size_t)E_ * D_ * F_ * 2;
  const size_t NEED    = W2B_OFF + (size_t)E_ * F_ * D_ * 2;
  const bool bf = (ws_size >= NEED);
  unsigned short* xbf  = (unsigned short*)(ws + XBF_OFF);
  unsigned short* w1bf = (unsigned short*)(ws + W1B_OFF);
  unsigned short* w2bf = (unsigned short*)(ws + W2B_OFF);

  init_meta_kernel<<<1, 32, 0, stream>>>(meta);
  proj_gate_kernel<<<NTOK / 8, 256, 0, stream>>>(x, vw, vb, gw, gb, v_ws, e_ws, meta);
  if (bf) {
    const int nx = NTOK * D_ / 2, nw = E_ * D_ * F_ / 2;
    cvt_bf16_kernel<<<(nx + 255) / 256, 256, 0, stream>>>(x,  (unsigned int*)xbf,  nx);
    cvt_bf16_kernel<<<(nw + 255) / 256, 256, 0, stream>>>(w1, (unsigned int*)w1bf, nw);
    cvt_bf16_kernel<<<(nw + 255) / 256, 256, 0, stream>>>(w2, (unsigned int*)w2bf, nw);
  }
  green_scan_kernel<<<1, 32, 0, stream>>>(v_ws, gamma, epsp, L_ws, U_ws);
  feats_kernel<<<NTOK / 256, 256, 0, stream>>>(v_ws, gamma, epsp, L_ws, U_ws, feats);
  scan_meta_kernel<<<1, 1, 0, stream>>>(meta);
  scatter_kernel<<<NTOK / 256, 256, 0, stream>>>(e_ws, meta, perm);
  // max tiles = sum ceil(count_e/32) <= 512 + 3; extra blocks exit immediately
  if (bf) {
    ffn_moe_kernel<true><<<NTOK / TMB + E_, 256, 0, stream>>>(
        x, xbf, w1, w1bf, b1, w2, w2bf, b2, ow, ob, bk, perm, meta, feats, out);
  } else {
    ffn_moe_kernel<false><<<NTOK / TMB + E_, 256, 0, stream>>>(
        x, xbf, w1, w1bf, b1, w2, w2bf, b2, ow, ob, bk, perm, meta, feats, out);
  }
}
